// aggregated_embedding_53437983096930
// MI455X (gfx1250) — matrix operand order repaired and checked by reading, not compiled, not run on hardware
//
#include <hip/hip_runtime.h>
#include <hip/hip_bf16.h>
#include <stdint.h>

// ---------------- CDNA5 WMMA types ----------------
typedef __attribute__((ext_vector_type(16))) __bf16 v16bf;
typedef __attribute__((ext_vector_type(8)))  float  v8f;

union Frag16 { uint4 u[2]; v16bf v; };
union Pack8  { unsigned short s[8]; uint4 u; };

__device__ __forceinline__ unsigned short f2bf(float f) {
  // round-to-nearest-even f32 -> bf16
  unsigned int u = __float_as_uint(f);
  return (unsigned short)((u + 0x7FFFu + ((u >> 16) & 1u)) >> 16);
}

#define D_MODEL 256
#define LDS_A_STRIDE 776  // bf16 elems/row: 768 + 8 pad (keeps 16B align, breaks bank stride)
#define LDS_E_STRIDE 264  // bf16 elems/row: 256 + 8 pad
#define LDS_R_STRIDE 260  // f32  elems/row: 256 + 4 pad

// ---------------- weight convert: out[n*C+k] = bf16(in[n*C+k]) ----------------
__global__ void conv_w_kernel(const float* __restrict__ in, unsigned short* __restrict__ out, int C) {
  int t = blockIdx.x * blockDim.x + threadIdx.x;
  if (t >= 256 * C) return;
  int n = t / C;
  int k = t - n * C;
  out[(size_t)n * C + k] = f2bf(in[t]);
}

// ---------------- leaves: leaves[b][j] = bf16(E[tokens[b, 2047+j]]) ----------------
__global__ void gather_leaves_kernel(const int* __restrict__ tokens,
                                     const float* __restrict__ E,
                                     unsigned short* __restrict__ leaves) {
  int t = blockIdx.x * blockDim.x + threadIdx.x;  // 64*2048*32 threads
  int l = t & 31;
  int row = t >> 5;                // b*2048 + j
  int b = row >> 11;
  int j = row & 2047;
  int tok = tokens[b * 4095 + 2047 + j];
  const float* erow = E + (size_t)tok * D_MODEL;
  float4 f0 = *(const float4*)(erow + l * 8);
  float4 f1 = *(const float4*)(erow + l * 8 + 4);
  Pack8 p;
  p.s[0] = f2bf(f0.x); p.s[1] = f2bf(f0.y); p.s[2] = f2bf(f0.z); p.s[3] = f2bf(f0.w);
  p.s[4] = f2bf(f1.x); p.s[5] = f2bf(f1.y); p.s[6] = f2bf(f1.z); p.s[7] = f2bf(f1.w);
  *(uint4*)(leaves + (size_t)row * D_MODEL + l * 8) = p.u;
}

// ---------------- one tree level: 16 nodes per block, 16 waves (one 16-wide N-tile each) ----
__global__ __launch_bounds__(512)
void level_kernel(const int* __restrict__ tokens,
                  const float* __restrict__ E,
                  const unsigned short* __restrict__ WtIn,  // [256][768] bf16 (= W_in, [n][k])
                  const unsigned short* __restrict__ Wt1,   // [256][256] bf16 (= W1, [n][k])
                  const unsigned short* __restrict__ Wt2,   // [256][256] bf16 (= W2, [n][k])
                  const float* __restrict__ b_in,
                  const float* __restrict__ b1,
                  const float* __restrict__ b2,
                  const unsigned short* __restrict__ prev,  // [64][2*level][256] bf16
                  unsigned short* __restrict__ next,        // [64][level][256] bf16
                  int lvlShift,                             // level = 1<<lvlShift
                  float* __restrict__ outF32)               // non-null only at final level
{
  __shared__ unsigned short aLds[16 * LDS_A_STRIDE];
  __shared__ unsigned short e1Lds[16 * LDS_E_STRIDE];
  __shared__ unsigned short e2Lds[16 * LDS_E_STRIDE];
  __shared__ float rootLds[16 * LDS_R_STRIDE];

  const int level = 1 << lvlShift;
  const int tid = threadIdx.x;
  const int g0 = blockIdx.x * 16;

  // ---- Stage A: build x tile (16 x 768 bf16) + f32 roots in LDS ----
  {
    const int m = tid >> 5;   // row 0..15
    const int l = tid & 31;   // 8-elem chunk 0..31
    const int g = g0 + m;
    const int b = g >> lvlShift;
    const int i = g & (level - 1);
    const int tok = tokens[b * 4095 + (level - 1) + i];
    const float* erow = E + (size_t)tok * D_MODEL;
    float4 f0 = *(const float4*)(erow + l * 8);
    float4 f1 = *(const float4*)(erow + l * 8 + 4);
    float* rr = rootLds + m * LDS_R_STRIDE + l * 8;
    *(float4*)(rr) = f0;
    *(float4*)(rr + 4) = f1;
    Pack8 p;
    p.s[0] = f2bf(f0.x); p.s[1] = f2bf(f0.y); p.s[2] = f2bf(f0.z); p.s[3] = f2bf(f0.w);
    p.s[4] = f2bf(f1.x); p.s[5] = f2bf(f1.y); p.s[6] = f2bf(f1.z); p.s[7] = f2bf(f1.w);
    *(uint4*)(aLds + m * LDS_A_STRIDE + l * 8) = p.u;
    // children (already bf16)
    const size_t base = (size_t)b * (2 * level) * D_MODEL;
    uint4 lv = *(const uint4*)(prev + base + (size_t)(2 * i)     * D_MODEL + l * 8);
    uint4 rv = *(const uint4*)(prev + base + (size_t)(2 * i + 1) * D_MODEL + l * 8);
    *(uint4*)(aLds + m * LDS_A_STRIDE + 256 + l * 8) = lv;
    *(uint4*)(aLds + m * LDS_A_STRIDE + 512 + l * 8) = rv;
  }
  __syncthreads();

  const int w    = tid >> 5;   // wave id == N-tile index (0..15)
  const int lane = tid & 31;
  const int hi   = lane >> 4;  // lane half
  const int nl   = lane & 15;
  const int mA   = nl;         // A-fragment row for this lane
  const int n    = w * 16 + nl;  // this lane's output column

  // ---- GEMM1: h1 = x @ W_in^T  (K = 768 -> 24 WMMA) ----
  v8f c1 = {};
  {
    const unsigned short* aBase = aLds + mA * LDS_A_STRIDE + 8 * hi;
#pragma unroll 4
    for (int kb = 0; kb < 24; ++kb) {
      Frag16 a, bf;
      a.u[0] = *(const uint4*)(aBase + kb * 32);
      a.u[1] = *(const uint4*)(aBase + kb * 32 + 16);
      const unsigned short* bp = WtIn + (size_t)n * 768 + kb * 32 + 8 * hi;
      bf.u[0] = *(const uint4*)(bp);
      bf.u[1] = *(const uint4*)(bp + 16);
      c1 = __builtin_amdgcn_wmma_f32_16x16x32_bf16(false, a.v, false, bf.v,
                                                   (short)0, c1, false, false);
    }
  }
  // epilogue 1: elu(h1 + b_in) -> e1 (bf16 in LDS)
  {
    const float bias = b_in[n];
#pragma unroll
    for (int r = 0; r < 8; ++r) {
      const int m = r + 8 * hi;
      float v = c1[r] + bias;
      v = (v > 0.0f) ? v : (__expf(v) - 1.0f);
      e1Lds[m * LDS_E_STRIDE + n] = f2bf(v);
    }
  }
  __syncthreads();

  // ---- GEMM2: h2 = e1 @ W1^T  (K = 256 -> 8 WMMA) ----
  v8f c2 = {};
  {
    const unsigned short* eBase = e1Lds + mA * LDS_E_STRIDE + 8 * hi;
#pragma unroll
    for (int kb = 0; kb < 8; ++kb) {
      Frag16 a, bf;
      a.u[0] = *(const uint4*)(eBase + kb * 32);
      a.u[1] = *(const uint4*)(eBase + kb * 32 + 16);
      const unsigned short* bp = Wt1 + (size_t)n * 256 + kb * 32 + 8 * hi;
      bf.u[0] = *(const uint4*)(bp);
      bf.u[1] = *(const uint4*)(bp + 16);
      c2 = __builtin_amdgcn_wmma_f32_16x16x32_bf16(false, a.v, false, bf.v,
                                                   (short)0, c2, false, false);
    }
  }
  // epilogue 2: elu(h2 + b1) -> e2
  {
    const float bias = b1[n];
#pragma unroll
    for (int r = 0; r < 8; ++r) {
      const int m = r + 8 * hi;
      float v = c2[r] + bias;
      v = (v > 0.0f) ? v : (__expf(v) - 1.0f);
      e2Lds[m * LDS_E_STRIDE + n] = f2bf(v);
    }
  }
  __syncthreads();

  // ---- GEMM3: h3 = e2 @ W2^T  (K = 256 -> 8 WMMA) ----
  v8f c3 = {};
  {
    const unsigned short* eBase = e2Lds + mA * LDS_E_STRIDE + 8 * hi;
#pragma unroll
    for (int kb = 0; kb < 8; ++kb) {
      Frag16 a, bf;
      a.u[0] = *(const uint4*)(eBase + kb * 32);
      a.u[1] = *(const uint4*)(eBase + kb * 32 + 16);
      const unsigned short* bp = Wt2 + (size_t)n * 256 + kb * 32 + 8 * hi;
      bf.u[0] = *(const uint4*)(bp);
      bf.u[1] = *(const uint4*)(bp + 16);
      c3 = __builtin_amdgcn_wmma_f32_16x16x32_bf16(false, a.v, false, bf.v,
                                                   (short)0, c3, false, false);
    }
  }
  // final epilogue: out = root + h3 + b2 -> bf16 cur (and f32 d_out on last level)
  {
    const float bias = b2[n];
#pragma unroll
    for (int r = 0; r < 8; ++r) {
      const int m = r + 8 * hi;
      const int g = g0 + m;
      float v = c3[r] + bias + rootLds[m * LDS_R_STRIDE + n];
      next[(size_t)g * D_MODEL + n] = f2bf(v);
      if (outF32) outF32[(size_t)g * D_MODEL + n] = v;
    }
  }
}

// ---------------- host launcher ----------------
extern "C" void kernel_launch(void* const* d_in, const int* in_sizes, int n_in,
                              void* d_out, int out_size, void* d_ws, size_t ws_size,
                              hipStream_t stream) {
  const int*   tokens = (const int*)  d_in[0];
  const float* E      = (const float*)d_in[1];
  const float* W_in   = (const float*)d_in[2];
  const float* b_in   = (const float*)d_in[3];
  const float* W1     = (const float*)d_in[4];
  const float* b1     = (const float*)d_in[5];
  const float* W2     = (const float*)d_in[6];
  const float* b2     = (const float*)d_in[7];

  char* ws = (char*)d_ws;
  unsigned short* WtIn = (unsigned short*)(ws);                       // 768*256*2 = 393216 B
  unsigned short* Wt1  = (unsigned short*)(ws + 393216);              // 131072 B
  unsigned short* Wt2  = (unsigned short*)(ws + 524288);              // 131072 B
  unsigned short* bufA = (unsigned short*)(ws + 655360);              // 64*2048*256*2 = 67108864 B
  unsigned short* bufB = (unsigned short*)(ws + 655360 + 67108864);   // 64*1024*256*2 = 33554432 B

  // 1) convert weights to bf16, keeping [n][k] layout
  conv_w_kernel<<<(256 * 768 + 255) / 256, 256, 0, stream>>>(W_in, WtIn, 768);
  conv_w_kernel<<<(256 * 256 + 255) / 256, 256, 0, stream>>>(W1, Wt1, 256);
  conv_w_kernel<<<(256 * 256 + 255) / 256, 256, 0, stream>>>(W2, Wt2, 256);

  // 2) gather leaf embeddings -> bf16
  gather_leaves_kernel<<<(64 * 2048 * 32) / 256, 256, 0, stream>>>(tokens, E, bufA);

  // 3) 11 tree levels, ping-ponging cur buffers
  unsigned short* prev = bufA;
  unsigned short* next = bufB;
  for (int shift = 10; shift >= 0; --shift) {
    const int level = 1 << shift;
    float* of = (shift == 0) ? (float*)d_out : nullptr;
    level_kernel<<<level * 4, 512, 0, stream>>>(tokens, E, WtIn, Wt1, Wt2,
                                                b_in, b1, b2, prev, next, shift, of);
    unsigned short* t = prev; prev = next; next = t;
  }
}
